// TranxParser_83502754169195
// MI455X (gfx1250) — matrix operand order repaired and checked by reading, not compiled, not run on hardware
//
#include <hip/hip_runtime.h>
#include <math.h>

typedef __bf16 bf16;
typedef __attribute__((ext_vector_type(16))) __bf16 v16bf;
typedef __attribute__((ext_vector_type(8)))  float  v8f;

// ---- problem dims ----
constexpr int cS  = 128;   // source length
constexpr int cB  = 1024;  // batch
constexpr int cT  = 96;    // decode steps
constexpr int cE  = 64;    // src emb
constexpr int cH  = 200;   // decoder hidden / ATT
constexpr int cHH = 100;   // per-dir encoder hidden
// padded GEMM dims
constexpr int KENC  = 192; // 64 emb + 100 h + 28 pad  (6 x 32)
constexpr int NENC  = 400; // 4*HH
constexpr int NENCP = 416; // Wenc padded rows (26 x 16, for 2x2 blocking)
constexpr int KDEC  = 640; // 440 inp + 200 h          (20 x 32)
constexpr int NDEC  = 800; // 4*H
constexpr int ENCP  = 224; // enc row stride (200 + pad, 7 x 32)
constexpr int EAP   = 208; // enc_att row stride (200 + pad, 13 x 16)
constexpr int NATTP = 224; // Wattn/Wav padded rows (14 x 16)
constexpr int KAV   = 416; // [ctx|h] K (400 + pad, 13 x 32)

#define CDIV(a,b) (((a)+(b)-1)/(b))

__device__ __forceinline__ float sigf(float x) { return 1.0f / (1.0f + __expf(-x)); }

// ---------------- WMMA fragment helpers (bf16, 16x16x32) ----------------
// A: row-major [M x K], 16-bit A layout: lane<16 -> row=lane, K = k0+{0..7,16..23}
//                                        lane>=16 -> row=lane-16, K = k0+{8..15,24..31}
__device__ __forceinline__ v16bf load_a_frag(const bf16* __restrict__ A, int lda,
                                             int m0, int k0, int lane) {
  int half = lane >> 4, m = lane & 15;
  const bf16* p = A + (size_t)(m0 + m) * lda + k0 + half * 8;
  v16bf a;
#pragma unroll
  for (int j = 0; j < 8; ++j) { a[j] = p[j]; a[8 + j] = p[16 + j]; }
  return a;
}
// B = W^T where W is row-major [N x K]; column n of B is row n of W.
// lanes 0-15: col=lane, K=k0+{0..7,16..23} ; lanes 16-31: col=lane-16, K=k0+{8..15,24..31}
__device__ __forceinline__ v16bf load_b_frag(const bf16* __restrict__ W, int ldw,
                                             int n0, int k0, int lane) {
  int half = lane >> 4, n = lane & 15;
  const bf16* p = W + (size_t)(n0 + n) * ldw + k0 + half * 8;
  v16bf b;
#pragma unroll
  for (int j = 0; j < 8; ++j) { b[j] = p[j]; b[8 + j] = p[16 + j]; }
  return b;
}

__device__ __forceinline__ void store_tile_f32(float* __restrict__ D, int ldd,
                                               int m0, int n0, v8f acc, int lane) {
  int half = lane >> 4, nn = lane & 15;
#pragma unroll
  for (int r = 0; r < 8; ++r)
    D[(size_t)(m0 + r + half * 8) * ldd + n0 + nn] = acc[r];
}
__device__ __forceinline__ void store_tile_bf16(bf16* __restrict__ D, int ldd,
                                                int m0, int n0, v8f acc, int lane) {
  int half = lane >> 4, nn = lane & 15;
#pragma unroll
  for (int r = 0; r < 8; ++r)
    D[(size_t)(m0 + r + half * 8) * ldd + n0 + nn] = (bf16)acc[r];
}

#define WMMA_BF16(a, b, c) \
  __builtin_amdgcn_wmma_f32_16x16x32_bf16(false, (a), false, (b), (short)0, (c), false, false)

// ------------- 2x2 register-blocked GEMM (one wave per 32x32 block) -------------
// W must be padded to Nb*32 rows; second N tile's store guarded by Nreal.
__global__ void k_gemm2_f32(const bf16* __restrict__ A, int lda,
                            const bf16* __restrict__ W, int ldw,
                            float* __restrict__ D, int ldd,
                            int Mb, int Nb, int K, int Nreal) {
  int wave = (blockIdx.x * blockDim.x + threadIdx.x) >> 5;
  int lane = threadIdx.x & 31;
  int bn = wave % Nb, bm = wave / Nb;
  if (bm >= Mb) return;                      // wave-uniform guard (EXEC stays all-1)
  int m0 = bm * 32, n0 = bn * 32;
  v8f acc00 = {}, acc01 = {}, acc10 = {}, acc11 = {};
  for (int k0 = 0; k0 < K; k0 += 32) {
    v16bf a0 = load_a_frag(A, lda, m0, k0, lane);
    v16bf a1 = load_a_frag(A, lda, m0 + 16, k0, lane);
    v16bf b0 = load_b_frag(W, ldw, n0, k0, lane);
    v16bf b1 = load_b_frag(W, ldw, n0 + 16, k0, lane);
    acc00 = WMMA_BF16(a0, b0, acc00);
    acc01 = WMMA_BF16(a0, b1, acc01);
    acc10 = WMMA_BF16(a1, b0, acc10);
    acc11 = WMMA_BF16(a1, b1, acc11);
  }
  store_tile_f32(D, ldd, m0, n0, acc00, lane);
  store_tile_f32(D, ldd, m0 + 16, n0, acc10, lane);
  if (n0 + 16 < Nreal) {
    store_tile_f32(D, ldd, m0, n0 + 16, acc01, lane);
    store_tile_f32(D, ldd, m0 + 16, n0 + 16, acc11, lane);
  }
}

__global__ void k_gemm2_bf16(const bf16* __restrict__ A, int lda,
                             const bf16* __restrict__ W, int ldw,
                             bf16* __restrict__ D, int ldd,
                             int Mb, int Nb, int K, int Nreal) {
  int wave = (blockIdx.x * blockDim.x + threadIdx.x) >> 5;
  int lane = threadIdx.x & 31;
  int bn = wave % Nb, bm = wave / Nb;
  if (bm >= Mb) return;
  int m0 = bm * 32, n0 = bn * 32;
  v8f acc00 = {}, acc01 = {}, acc10 = {}, acc11 = {};
  for (int k0 = 0; k0 < K; k0 += 32) {
    v16bf a0 = load_a_frag(A, lda, m0, k0, lane);
    v16bf a1 = load_a_frag(A, lda, m0 + 16, k0, lane);
    v16bf b0 = load_b_frag(W, ldw, n0, k0, lane);
    v16bf b1 = load_b_frag(W, ldw, n0 + 16, k0, lane);
    acc00 = WMMA_BF16(a0, b0, acc00);
    acc01 = WMMA_BF16(a0, b1, acc01);
    acc10 = WMMA_BF16(a1, b0, acc10);
    acc11 = WMMA_BF16(a1, b1, acc11);
  }
  store_tile_bf16(D, ldd, m0, n0, acc00, lane);
  store_tile_bf16(D, ldd, m0 + 16, n0, acc10, lane);
  if (n0 + 16 < Nreal) {
    store_tile_bf16(D, ldd, m0, n0 + 16, acc01, lane);
    store_tile_bf16(D, ldd, m0 + 16, n0 + 16, acc11, lane);
  }
}

// s_att = tanh([ctx|h] @ W_av^T); dual store: fp32 out slice + bf16 into X[:,32..231]
__device__ __forceinline__ void store_satt(float* __restrict__ out_t,
                                           bf16* __restrict__ X,
                                           int m0, int n0, v8f acc, int lane) {
  int half = lane >> 4, nn = lane & 15;
  int col = n0 + nn;
  if (col < cH) {
#pragma unroll
    for (int r = 0; r < 8; ++r) {
      int row = m0 + r + half * 8;
      float v = tanhf(acc[r]);
      out_t[(size_t)row * cH + col] = v;
      X[(size_t)row * KDEC + 32 + col] = (bf16)v;
    }
  }
}
__global__ void k_gemm2_satt(const bf16* __restrict__ A, int lda,
                             const bf16* __restrict__ W, int ldw,
                             float* __restrict__ out_t, bf16* __restrict__ X,
                             int Mb, int Nb, int K) {
  int wave = (blockIdx.x * blockDim.x + threadIdx.x) >> 5;
  int lane = threadIdx.x & 31;
  int bn = wave % Nb, bm = wave / Nb;
  if (bm >= Mb) return;
  int m0 = bm * 32, n0 = bn * 32;
  v8f acc00 = {}, acc01 = {}, acc10 = {}, acc11 = {};
  for (int k0 = 0; k0 < K; k0 += 32) {
    v16bf a0 = load_a_frag(A, lda, m0, k0, lane);
    v16bf a1 = load_a_frag(A, lda, m0 + 16, k0, lane);
    v16bf b0 = load_b_frag(W, ldw, n0, k0, lane);
    v16bf b1 = load_b_frag(W, ldw, n0 + 16, k0, lane);
    acc00 = WMMA_BF16(a0, b0, acc00);
    acc01 = WMMA_BF16(a0, b1, acc01);
    acc10 = WMMA_BF16(a1, b0, acc10);
    acc11 = WMMA_BF16(a1, b1, acc11);
  }
  store_satt(out_t, X, m0, n0, acc00, lane);
  store_satt(out_t, X, m0 + 16, n0, acc10, lane);
  store_satt(out_t, X, m0, n0 + 16, acc01, lane);
  store_satt(out_t, X, m0 + 16, n0 + 16, acc11, lane);
}

// ---------------- weight packing ----------------
__global__ void k_pack_enc(const float* Wih_f, const float* Whh_f,
                           const float* Wih_b, const float* Whh_b, bf16* Wenc) {
  int idx = blockIdx.x * blockDim.x + threadIdx.x;
  if (idx >= 2 * NENCP * KENC) return;
  int dir = idx / (NENCP * KENC);
  int r = idx % (NENCP * KENC);
  int n = r / KENC, k = r % KENC;
  const float* Wih = dir ? Wih_b : Wih_f;
  const float* Whh = dir ? Whh_b : Whh_f;
  float v = 0.0f;
  if (n < NENC) {
    v = (k < cE) ? Wih[n * cE + k]
      : (k < cE + cHH) ? Whh[n * cHH + (k - cE)] : 0.0f;
  }
  Wenc[idx] = (bf16)v;
}
__global__ void k_pack_dec(const float* Wih_d, const float* Whh_d, bf16* Wcat) {
  int idx = blockIdx.x * blockDim.x + threadIdx.x;
  if (idx >= NDEC * KDEC) return;
  int n = idx / KDEC, k = idx % KDEC;
  float v = (k < 440) ? Wih_d[n * 440 + k] : Whh_d[n * cH + (k - 440)];
  Wcat[idx] = (bf16)v;
}
__global__ void k_pack_attn(const float* W_attn, bf16* Wa) {
  int idx = blockIdx.x * blockDim.x + threadIdx.x;
  if (idx >= NATTP * ENCP) return;
  int n = idx / ENCP, k = idx % ENCP;
  Wa[idx] = (bf16)((n < cH && k < cH) ? W_attn[n * cH + k] : 0.0f);
}
__global__ void k_pack_av(const float* W_av, bf16* Wv) {
  int idx = blockIdx.x * blockDim.x + threadIdx.x;
  if (idx >= NATTP * KAV) return;
  int n = idx / KAV, k = idx % KAV;
  Wv[idx] = (bf16)((n < cH && k < 2 * cH) ? W_av[n * 2 * cH + k] : 0.0f);
}

// zero c_enc and the enc pad columns (200..223)
__global__ void k_setup_zero(float* c_enc, bf16* enc) {
  int idx = blockIdx.x * blockDim.x + threadIdx.x;
  int n1 = 2 * cB * cHH;
  int n2 = cB * cS * (ENCP - cH);
  if (idx < n1) { c_enc[idx] = 0.0f; return; }
  idx -= n1;
  if (idx >= n2) return;
  int bs = idx / (ENCP - cH), j = idx % (ENCP - cH);
  enc[(size_t)bs * ENCP + cH + j] = (bf16)0.0f;
}

// fill Xf/Xb: emb cols 0..63 for every step; zero pad cols 164..191 everywhere;
// zero h cols 64..163 for step-index 0 only (later steps written by enc cell).
__global__ void k_gather_emb(const int* sentences, const float* src_emb,
                             bf16* Xf, bf16* Xb) {
  long long idx = (long long)blockIdx.x * blockDim.x + threadIdx.x;
  long long total = 2LL * cS * cB * KENC;
  if (idx >= total) return;
  int dir = (int)(idx / ((long long)cS * cB * KENC));
  long long r = idx % ((long long)cS * cB * KENC);
  int srow = (int)(r / (cB * KENC));
  int rem = (int)(r % (cB * KENC));
  int b = rem / KENC, col = rem % KENC;
  int s_act = dir ? (cS - 1 - srow) : srow;
  bf16* X = (dir ? Xb : Xf) + ((size_t)srow * cB + b) * KENC;
  if (col < cE) {
    int tok = sentences[s_act * cB + b];
    X[col] = (bf16)src_emb[(size_t)tok * cE + col];
  } else if (col >= cE + cHH) {
    X[col] = (bf16)0.0f;
  } else if (srow == 0) {
    X[col] = (bf16)0.0f;
  }
}

// encoder LSTM cell for both directions at processed-step k
__global__ void k_enc_cell(const float* z_enc, const float* b_f, const float* b_b,
                           float* c_enc, bf16* Xf, bf16* Xb, bf16* enc,
                           float* hT, int k) {
  int idx = blockIdx.x * blockDim.x + threadIdx.x;
  if (idx >= 2 * cB * cHH) return;
  int dir = idx / (cB * cHH);
  int r = idx % (cB * cHH);
  int b = r / cHH, j = r % cHH;
  const float* z = z_enc + (size_t)dir * cB * NENC + (size_t)b * NENC;
  const float* bias = dir ? b_b : b_f;
  float gi = z[j]          + bias[j];
  float gf = z[cHH + j]    + bias[cHH + j];
  float gg = z[2*cHH + j]  + bias[2*cHH + j];
  float go = z[3*cHH + j]  + bias[3*cHH + j];
  size_t ci = (size_t)dir * cB * cHH + (size_t)b * cHH + j;
  float c = c_enc[ci];
  c = sigf(gf) * c + sigf(gi) * tanhf(gg);
  float h = sigf(go) * tanhf(c);
  c_enc[ci] = c;
  hT[ci] = h;
  int s_act = dir ? (cS - 1 - k) : k;
  enc[((size_t)b * cS + s_act) * ENCP + dir * cHH + j] = (bf16)h;
  if (k + 1 < cS) {
    bf16* Xn = (dir ? Xb : Xf) + ((size_t)(k + 1) * cB + b) * KENC;
    Xn[cE + j] = (bf16)h;
  }
}

// decoder init: torch-style h0/c0 reshape [2,B,HH] -> [B,200]; zero X inp + X2 pad
__global__ void k_dec_init(const float* hT, const float* cT, float* c_dec,
                           bf16* h200, bf16* X, bf16* X2) {
  int idx = blockIdx.x * blockDim.x + threadIdx.x;
  if (idx >= cB * 656) return;
  int b = idx / 656, c = idx % 656;
  if (c < cH) {
    int flat = b * cH + c;
    int d = flat / (cB * cHH);
    int rem = flat % (cB * cHH);
    int bb = rem / cHH, kk = rem % cHH;
    size_t si = (size_t)d * cB * cHH + (size_t)bb * cHH + kk;
    float h0 = hT[si];
    c_dec[(size_t)b * cH + c] = cT[si];
    bf16 hb = (bf16)h0;
    h200[(size_t)b * cH + c] = hb;
    X[(size_t)b * KDEC + 440 + c] = hb;
  } else if (c < 640) {
    X[(size_t)b * KDEC + (c - 200)] = (bf16)0.0f;   // zero inp cols 0..439 (t==0)
  } else {
    X2[(size_t)b * KAV + 400 + (c - 640)] = (bf16)0.0f;
  }
}

// per-step gather (t>=1): a_tm1 (cols 0..31), field (232..239), parent (240..439)
__global__ void k_gather(const int* is_prod, const int* prod_ids, const int* prim_ids,
                         const int* field_ids, const int* parent_t,
                         const float* actprod_emb, const float* prim_emb,
                         const float* field_emb, const bf16* Hbuf, bf16* X, int t) {
  int idx = blockIdx.x * blockDim.x + threadIdx.x;
  if (idx >= cB * 240) return;
  int b = idx / 240, c = idx % 240;
  if (c < 32) {
    int ip = is_prod[t * cB + b];
    float v = (ip == 1) ? actprod_emb[prod_ids[t * cB + b] * 32 + c]
                        : prim_emb[(size_t)prim_ids[t * cB + b] * 32 + c];
    X[(size_t)b * KDEC + c] = (bf16)v;
  } else {
    int col = c + 200;                       // 232..439
    if (col < 240) {
      float v = field_emb[field_ids[t * cB + b] * 8 + (col - 232)];
      X[(size_t)b * KDEC + col] = (bf16)v;
    } else {
      int pt = parent_t[t * cB + b];
      X[(size_t)b * KDEC + col] = Hbuf[((size_t)pt * cB + b) * cH + (col - 240)];
    }
  }
}

__global__ void k_dec_cell(const float* z, const float* bd, float* c,
                           bf16* X, bf16* h200, bf16* Hbuf, int t) {
  int idx = blockIdx.x * blockDim.x + threadIdx.x;
  if (idx >= cB * cH) return;
  int b = idx / cH, j = idx % cH;
  const float* zr = z + (size_t)b * NDEC;
  float gi = zr[j]         + bd[j];
  float gf = zr[cH + j]    + bd[cH + j];
  float gg = zr[2*cH + j]  + bd[2*cH + j];
  float go = zr[3*cH + j]  + bd[3*cH + j];
  float cv = c[idx];
  cv = sigf(gf) * cv + sigf(gi) * tanhf(gg);
  float h = sigf(go) * tanhf(cv);
  c[idx] = cv;
  bf16 hb = (bf16)h;
  h200[idx] = hb;
  X[(size_t)b * KDEC + 440 + j] = hb;
  Hbuf[((size_t)t * cB + b) * cH + j] = hb;
}

// attention: scores -> softmax -> ctx ; X2 = [ctx | h] (bf16, K-padded)
__global__ void k_attention(const bf16* __restrict__ enc_att,
                            const bf16* __restrict__ enc,
                            const bf16* __restrict__ h200, bf16* __restrict__ X2) {
  int b = blockIdx.x, tid = threadIdx.x;     // blockDim = 128 = cS
  __shared__ float sm[cS];
  __shared__ float bc;
  const bf16* hb = h200 + (size_t)b * cH;
  {
    const bf16* ea = enc_att + ((size_t)b * cS + tid) * EAP;
    float acc = 0.0f;
    for (int j = 0; j < cH; ++j) acc += (float)ea[j] * (float)hb[j];
    sm[tid] = acc;
  }
  __syncthreads();
  if (tid == 0) {
    float m = sm[0];
    for (int s = 1; s < cS; ++s) m = fmaxf(m, sm[s]);
    bc = m;
  }
  __syncthreads();
  float e = __expf(sm[tid] - bc);
  __syncthreads();
  sm[tid] = e;
  __syncthreads();
  if (tid == 0) {
    float s0 = 0.0f;
    for (int s = 0; s < cS; ++s) s0 += sm[s];
    bc = 1.0f / s0;
  }
  __syncthreads();
  sm[tid] = sm[tid] * bc;
  __syncthreads();
  for (int j = tid; j < cH; j += cS) {
    const bf16* eb = enc + (size_t)b * cS * ENCP + j;
    float acc = 0.0f;
    for (int s = 0; s < cS; ++s) acc += sm[s] * (float)eb[(size_t)s * ENCP];
    X2[(size_t)b * KAV + j] = (bf16)acc;
    X2[(size_t)b * KAV + cH + j] = hb[j];
  }
}

extern "C" void kernel_launch(void* const* d_in, const int* in_sizes, int n_in,
                              void* d_out, int out_size, void* d_ws, size_t ws_size,
                              hipStream_t stream) {
  (void)in_sizes; (void)n_in; (void)out_size; (void)ws_size;
  const int*   sentences   = (const int*)  d_in[0];
  const int*   is_prod     = (const int*)  d_in[1];
  const int*   prod_ids    = (const int*)  d_in[2];
  const int*   prim_ids    = (const int*)  d_in[3];
  const int*   field_ids   = (const int*)  d_in[4];
  const int*   parent_t    = (const int*)  d_in[5];
  const float* src_emb     = (const float*)d_in[6];
  const float* actprod_emb = (const float*)d_in[7];
  const float* prim_emb    = (const float*)d_in[8];
  const float* field_emb   = (const float*)d_in[9];
  const float* Wih_f = (const float*)d_in[10];
  const float* Whh_f = (const float*)d_in[11];
  const float* b_f   = (const float*)d_in[12];
  const float* Wih_b = (const float*)d_in[13];
  const float* Whh_b = (const float*)d_in[14];
  const float* b_b   = (const float*)d_in[15];
  const float* Wih_d = (const float*)d_in[16];
  const float* Whh_d = (const float*)d_in[17];
  const float* b_d   = (const float*)d_in[18];
  const float* W_attn= (const float*)d_in[19];
  const float* W_av  = (const float*)d_in[20];
  float* out = (float*)d_out;

  // ---- workspace carve-up (256B aligned) ----
  char* ws = (char*)d_ws;
  size_t off = 0;
  auto alloc = [&](size_t bytes) -> char* {
    char* p = ws + off;
    off += (bytes + 255) & ~(size_t)255;
    return p;
  };
  bf16*  Xf     = (bf16*) alloc((size_t)cS * cB * KENC * 2);
  bf16*  Xb     = (bf16*) alloc((size_t)cS * cB * KENC * 2);
  bf16*  Wenc   = (bf16*) alloc((size_t)2 * NENCP * KENC * 2);
  bf16*  Wcat   = (bf16*) alloc((size_t)NDEC * KDEC * 2);
  bf16*  Wattn  = (bf16*) alloc((size_t)NATTP * ENCP * 2);
  bf16*  Wav    = (bf16*) alloc((size_t)NATTP * KAV * 2);
  float* z_enc  = (float*)alloc((size_t)2 * cB * NENC * 4);
  float* c_enc  = (float*)alloc((size_t)2 * cB * cHH * 4);
  float* hT     = (float*)alloc((size_t)2 * cB * cHH * 4);
  bf16*  enc    = (bf16*) alloc((size_t)cB * cS * ENCP * 2);
  bf16*  encatt = (bf16*) alloc((size_t)cB * cS * EAP * 2);
  bf16*  Xd     = (bf16*) alloc((size_t)cB * KDEC * 2);
  float* z_dec  = (float*)alloc((size_t)cB * NDEC * 4);
  float* c_dec  = (float*)alloc((size_t)cB * cH * 4);
  bf16*  h200   = (bf16*) alloc((size_t)cB * cH * 2);
  bf16*  Hbuf   = (bf16*) alloc((size_t)cT * cB * cH * 2);
  bf16*  X2     = (bf16*) alloc((size_t)cB * KAV * 2);

  // ---- setup / packing ----
  k_pack_enc <<<CDIV(2 * NENCP * KENC, 256), 256, 0, stream>>>(Wih_f, Whh_f, Wih_b, Whh_b, Wenc);
  k_pack_dec <<<CDIV(NDEC * KDEC, 256), 256, 0, stream>>>(Wih_d, Whh_d, Wcat);
  k_pack_attn<<<CDIV(NATTP * ENCP, 256), 256, 0, stream>>>(W_attn, Wattn);
  k_pack_av  <<<CDIV(NATTP * KAV, 256), 256, 0, stream>>>(W_av, Wav);
  k_setup_zero<<<CDIV(2 * cB * cHH + cB * cS * (ENCP - cH), 256), 256, 0, stream>>>(c_enc, enc);
  {
    long long tot = 2LL * cS * cB * KENC;
    k_gather_emb<<<(unsigned)CDIV(tot, 256), 256, 0, stream>>>(sentences, src_emb, Xf, Xb);
  }

  // ---- encoder recurrence (128 sequential steps, fwd + bwd) ----
  // 2x2 blocking: Mb = 1024/32 = 32, Nb = 416/32 = 13 (last half-block store-guarded)
  const int encBlocks = CDIV(32 * 13 * 32, 256);
  for (int k = 0; k < cS; ++k) {
    k_gemm2_f32<<<encBlocks, 256, 0, stream>>>(Xf + (size_t)k * cB * KENC, KENC,
                                               Wenc, KENC, z_enc, NENC,
                                               32, 13, KENC, NENC);
    k_gemm2_f32<<<encBlocks, 256, 0, stream>>>(Xb + (size_t)k * cB * KENC, KENC,
                                               Wenc + (size_t)NENCP * KENC, KENC,
                                               z_enc + (size_t)cB * NENC, NENC,
                                               32, 13, KENC, NENC);
    k_enc_cell<<<CDIV(2 * cB * cHH, 256), 256, 0, stream>>>(z_enc, b_f, b_b, c_enc,
                                                            Xf, Xb, enc, hT, k);
  }

  // ---- enc_att = enc @ W_attn^T  (big parallel bf16 WMMA GEMM) ----
  {
    int Mb = (cB * cS) / 32, Nb = NATTP / 32;     // 4096 x 7
    int blocks = CDIV(Mb * Nb * 32, 256);
    k_gemm2_bf16<<<blocks, 256, 0, stream>>>(enc, ENCP, Wattn, ENCP,
                                             encatt, EAP, Mb, Nb, ENCP, EAP);
  }

  // ---- decoder init ----
  k_dec_init<<<CDIV(cB * 656, 256), 256, 0, stream>>>(hT, c_enc, c_dec, h200, Xd, X2);

  // ---- decoder recurrence (96 sequential steps) ----
  const int decBlocks = CDIV(32 * 25 * 32, 256);  // Mb=32, Nb=25 (800/32)
  const int savBlocks = CDIV(32 * 7 * 32, 256);   // Mb=32, Nb=7  (224/32)
  for (int t = 0; t < cT; ++t) {
    if (t > 0)
      k_gather<<<CDIV(cB * 240, 256), 256, 0, stream>>>(is_prod, prod_ids, prim_ids,
                                                        field_ids, parent_t,
                                                        actprod_emb, prim_emb, field_emb,
                                                        Hbuf, Xd, t);
    k_gemm2_f32<<<decBlocks, 256, 0, stream>>>(Xd, KDEC, Wcat, KDEC, z_dec, NDEC,
                                               32, 25, KDEC, NDEC);
    k_dec_cell<<<CDIV(cB * cH, 256), 256, 0, stream>>>(z_dec, b_d, c_dec, Xd, h200, Hbuf, t);
    k_attention<<<cB, cS, 0, stream>>>(encatt, enc, h200, X2);
    k_gemm2_satt<<<savBlocks, 256, 0, stream>>>(X2, KAV, Wav, KAV,
                                                out + (size_t)t * cB * cH, Xd,
                                                32, 7, KAV);
  }
}
